// GroupedQueryAttention_74680891343279
// MI455X (gfx1250) — matrix operand order repaired and checked by reading, not compiled, not run on hardware
//
#include <hip/hip_runtime.h>
#include <hip/hip_bf16.h>

typedef __attribute__((ext_vector_type(16))) _Float16 v16h;
typedef __attribute__((ext_vector_type(8)))  _Float16 v8h;
typedef __attribute__((ext_vector_type(4)))  _Float16 v4h;
typedef __attribute__((ext_vector_type(8)))  float    v8f;
typedef __attribute__((ext_vector_type(4)))  float    v4f;

#define T_SEQ   4096
#define D_MODEL 2048
#define D_KV    512
#define NQH     16
#define NKVH    4
#define D_HEAD  128

static __device__ inline v16h cat8(v8h lo, v8h hi) {
  return __builtin_shufflevector(lo, hi, 0,1,2,3,4,5,6,7,8,9,10,11,12,13,14,15);
}

// ---- element-type helpers for the templated GEMM ----
static __device__ inline void ld4h(const float* p, _Float16* d) {
  v4f t = *(const v4f*)p;
  d[0]=(_Float16)t[0]; d[1]=(_Float16)t[1]; d[2]=(_Float16)t[2]; d[3]=(_Float16)t[3];
}
static __device__ inline void ld4h(const _Float16* p, _Float16* d) {
  v4h t = *(const v4h*)p;
  d[0]=t[0]; d[1]=t[1]; d[2]=t[2]; d[3]=t[3];
}
static __device__ inline void stc(float v, float* p)    { *p = v; }
static __device__ inline void stc(float v, _Float16* p) { *p = (_Float16)v; }

// =====================================================================
// Tiled WMMA GEMM: C[MxN] = A[MxK] @ B[KxN]; A = f32 or f16, B = f32,
// C = f32 or f16. Block tile 128x128, BK=32, 8 waves (2 M x 4 N),
// per-wave tile 64x32 = 4x2 WMMA accumulators.
// =====================================================================
template<typename AT, typename CT>
__global__ __launch_bounds__(256) void gemm_wmma_kernel(
    const AT* __restrict__ A, const float* __restrict__ B, CT* __restrict__ C,
    int M, int N, int K)
{
  __shared__ __align__(16) _Float16 As[128 * 32];   // [m][k]
  __shared__ __align__(16) _Float16 Bs[128 * 32];   // [n][k] (transposed tile)
  const int tid   = threadIdx.x;
  const int lane  = tid & 31;
  const int wave  = tid >> 5;
  const int mW    = wave >> 2;           // 0..1
  const int nW    = wave & 3;            // 0..3
  const int mBase = blockIdx.y * 128;
  const int nBase = blockIdx.x * 128;
  const int lm    = lane & 15;
  const int hiH   = (lane < 16) ? 0 : 1;

  v8f zc = {};
  v8f acc[4][2];
#pragma unroll
  for (int i = 0; i < 4; ++i) { acc[i][0] = zc; acc[i][1] = zc; }

  for (int k0 = 0; k0 < K; k0 += 32) {
    // ---- cooperative stage of A(128x32) and B(32x128, stored [n][k]) ----
#pragma unroll
    for (int i = 0; i < 4; ++i) {
      int g  = tid + i * 256;            // 0..1023 groups of 4 elements
      int ar = g >> 3;                   // A row 0..127
      int ac = (g & 7) * 4;              // A col 0..28
      _Float16 tmp[4];
      ld4h(A + (size_t)(mBase + ar) * K + k0 + ac, tmp);
      As[ar*32 + ac+0] = tmp[0]; As[ar*32 + ac+1] = tmp[1];
      As[ar*32 + ac+2] = tmp[2]; As[ar*32 + ac+3] = tmp[3];

      int br = g >> 5;                   // B k-row 0..31
      int bc = (g & 31) * 4;             // B n-col 0..124
      v4f bt = *(const v4f*)(B + (size_t)(k0 + br) * N + nBase + bc);
      Bs[(bc+0)*32 + br] = (_Float16)bt[0];
      Bs[(bc+1)*32 + br] = (_Float16)bt[1];
      Bs[(bc+2)*32 + br] = (_Float16)bt[2];
      Bs[(bc+3)*32 + br] = (_Float16)bt[3];
    }
    __syncthreads();

    if (k0 + 32 < K) {  // prefetch next tiles into cache (global_prefetch_b8)
      __builtin_prefetch(A + (size_t)(mBase + (tid >> 1)) * K + k0 + 32 + (tid & 1) * 16, 0, 1);
      __builtin_prefetch(B + (size_t)(k0 + 32 + (tid >> 3)) * N + nBase + (tid & 7) * 16, 0, 1);
    }

    // ---- B fragments: lanes 0-15 hold K=0..15, lanes 16-31 hold K=16..31 ----
    v16h bf[2];
#pragma unroll
    for (int nt = 0; nt < 2; ++nt) {
      const _Float16* p = &Bs[(nW*32 + nt*16 + lm) * 32 + (hiH ? 16 : 0)];
      bf[nt] = cat8(*(const v8h*)p, *(const v8h*)(p + 8));
    }
    // ---- A fragments: lanes 0-15 hold K=0..15, lanes 16-31 hold K=16..31 ----
#pragma unroll
    for (int mt = 0; mt < 4; ++mt) {
      const _Float16* p = &As[(mW*64 + mt*16 + lm) * 32 + (hiH ? 16 : 0)];
      v16h af = cat8(*(const v8h*)p, *(const v8h*)(p + 8));
#pragma unroll
      for (int nt = 0; nt < 2; ++nt)
        acc[mt][nt] = __builtin_amdgcn_wmma_f32_16x16x32_f16(
            false, af, false, bf[nt], (short)0, acc[mt][nt], false, false);
    }
    __syncthreads();
  }

  // ---- store C: VGPR j -> row j (lanes 0-15) / j+8 (lanes 16-31), col = lane%16 ----
#pragma unroll
  for (int mt = 0; mt < 4; ++mt)
#pragma unroll
    for (int nt = 0; nt < 2; ++nt)
#pragma unroll
      for (int j = 0; j < 8; ++j) {
        int row = mBase + mW*64 + mt*16 + j + hiH*8;
        int col = nBase + nW*32 + nt*16 + lm;
        stc(acc[mt][nt][j], C + (size_t)row * N + col);
      }
}

// =====================================================================
// RoPE (interleaved pairs within each head's 128 dims), in-place on f16.
// =====================================================================
__global__ __launch_bounds__(256) void rope_kernel(_Float16* X, int nHeads)
{
  int tid  = blockIdx.x * 256 + threadIdx.x;
  int perT = nHeads * 64;
  int t = tid / perT;
  int r = tid % perT;
  int h = r >> 6;
  int i = r & 63;
  float ang = (float)t * __powf(10000.0f, -(float)i * (1.0f / 64.0f));
  float s, c;
  __sincosf(ang, &s, &c);
  _Float16* p = X + (size_t)t * ((size_t)nHeads * D_HEAD) + h * D_HEAD + 2 * i;
  float x0 = (float)p[0], x1 = (float)p[1];
  p[0] = (_Float16)(x0 * c - x1 * s);
  p[1] = (_Float16)(x0 * s + x1 * c);
}

// =====================================================================
// V transpose: [T][512] -> dim-major [512][T] so PV B-fragments are
// contiguous loads.
// =====================================================================
__global__ __launch_bounds__(256) void transpose_v_kernel(
    const _Float16* __restrict__ V, _Float16* __restrict__ Vt)
{
  int e = blockIdx.x * 256 + threadIdx.x;  // T*512 elements
  int t = e >> 9;
  int c = e & 511;
  Vt[(size_t)c * T_SEQ + t] = V[e];
}

// =====================================================================
// FlashAttention: one wave = one (head, 16-row query block). Streams 32
// keys per step: S = Q K^T (8 WMMA), online causal softmax, O += P V
// (8 WMMA). P re-layout C->A via per-wave LDS tile.
// =====================================================================
__global__ __launch_bounds__(256) void flash_attn_kernel(
    const _Float16* __restrict__ Q, const _Float16* __restrict__ Km,
    const _Float16* __restrict__ Vt, _Float16* __restrict__ ctx)
{
  __shared__ __align__(16) _Float16 Plds[8][16 * 32];
  const int lane = threadIdx.x & 31;
  const int wave = threadIdx.x >> 5;
  const int W    = blockIdx.x * 8 + wave;    // 0..4095
  const int head = W >> 8;                   // 0..15
  const int qb   = W & 255;                  // 0..255
  const int hkv  = head >> 2;                // GQA: 4 Q-heads per KV head
  const int qRowBase = qb * 16;
  const int lm   = lane & 15;
  const int hiH  = (lane < 16) ? 0 : 1;
  const float isq = 0.088388347648318447f;   // 1/sqrt(128)

  // ---- preload Q fragments for all 4 K-steps of d_head=128 ----
  v16h qf[4];
#pragma unroll
  for (int ks = 0; ks < 4; ++ks) {
    const _Float16* p = Q + (size_t)(qRowBase + lm) * D_MODEL
                          + head * D_HEAD + ks * 32 + (hiH ? 16 : 0);
    qf[ks] = cat8(*(const v8h*)p, *(const v8h*)(p + 8));
  }

  v8f zc = {};
  v8f o[8];
#pragma unroll
  for (int t = 0; t < 8; ++t) o[t] = zc;
  float mrow[8], lrow[8];
#pragma unroll
  for (int j = 0; j < 8; ++j) { mrow[j] = -3.0e38f; lrow[j] = 0.0f; }

  _Float16* P = &Plds[wave][0];

  for (int kb = 0; kb < qRowBase + 16; kb += 32) {
    // ---- S = Q K^T for 32 keys (two 16x16 C tiles) ----
    v8f s[2]; s[0] = zc; s[1] = zc;
#pragma unroll
    for (int nt = 0; nt < 2; ++nt) {
#pragma unroll
      for (int ks = 0; ks < 4; ++ks) {
        const _Float16* p = Km + (size_t)(kb + nt*16 + lm) * D_KV
                               + hkv * D_HEAD + ks * 32 + (hiH ? 16 : 0);
        v16h kf = cat8(*(const v8h*)p, *(const v8h*)(p + 8));
        s[nt] = __builtin_amdgcn_wmma_f32_16x16x32_f16(
            false, qf[ks], false, kf, (short)0, s[nt], false, false);
      }
    }

    // ---- scale, causal mask, online softmax stats ----
    float corr[8];
#pragma unroll
    for (int j = 0; j < 8; ++j) {
      int qRow = qRowBase + j + hiH * 8;
#pragma unroll
      for (int nt = 0; nt < 2; ++nt) {
        int key = kb + nt * 16 + lm;
        float v = s[nt][j] * isq;
        s[nt][j] = (key > qRow) ? -1.0e30f : v;
      }
      float mj = fmaxf(s[0][j], s[1][j]);
#pragma unroll
      for (int off = 8; off >= 1; off >>= 1) mj = fmaxf(mj, __shfl_xor(mj, off, 16));
      float mn = fmaxf(mrow[j], mj);
      corr[j] = __expf(mrow[j] - mn);
      s[0][j] = __expf(s[0][j] - mn);
      s[1][j] = __expf(s[1][j] - mn);
      float rs = s[0][j] + s[1][j];
#pragma unroll
      for (int off = 8; off >= 1; off >>= 1) rs += __shfl_xor(rs, off, 16);
      lrow[j] = lrow[j] * corr[j] + rs;
      mrow[j] = mn;
    }
#pragma unroll
    for (int t = 0; t < 8; ++t)
#pragma unroll
      for (int j = 0; j < 8; ++j) o[t][j] *= corr[j];

    // ---- P (C layout) -> LDS row-major 16x32 f16 ----
#pragma unroll
    for (int nt = 0; nt < 2; ++nt)
#pragma unroll
      for (int j = 0; j < 8; ++j)
        P[(j + hiH*8) * 32 + nt*16 + lm] = (_Float16)s[nt][j];
    asm volatile("s_wait_dscnt 0" ::: "memory");

    // ---- O += P @ V ----
    {
      const _Float16* pp = &P[lm * 32 + (hiH ? 16 : 0)];
      v16h pf = cat8(*(const v8h*)pp, *(const v8h*)(pp + 8));
#pragma unroll
      for (int t = 0; t < 8; ++t) {
        const _Float16* vp = Vt + (size_t)(hkv * D_HEAD + t*16 + lm) * T_SEQ
                                + kb + (hiH ? 16 : 0);
        v16h vf = cat8(*(const v8h*)vp, *(const v8h*)(vp + 8));
        o[t] = __builtin_amdgcn_wmma_f32_16x16x32_f16(
            false, pf, false, vf, (short)0, o[t], false, false);
      }
    }
  }

  // ---- epilogue: normalize and store ctx (f16) ----
#pragma unroll
  for (int j = 0; j < 8; ++j) lrow[j] = 1.0f / lrow[j];
#pragma unroll
  for (int t = 0; t < 8; ++t)
#pragma unroll
    for (int j = 0; j < 8; ++j) {
      int row = qRowBase + j + hiH * 8;
      int col = head * D_HEAD + t * 16 + lm;
      ctx[(size_t)row * D_MODEL + col] = (_Float16)(o[t][j] * lrow[j]);
    }
}

// =====================================================================
extern "C" void kernel_launch(void* const* d_in, const int* in_sizes, int n_in,
                              void* d_out, int out_size, void* d_ws, size_t ws_size,
                              hipStream_t stream)
{
  const float* x  = (const float*)d_in[0];
  const float* WQ = (const float*)d_in[1];
  const float* WK = (const float*)d_in[2];
  const float* WV = (const float*)d_in[3];
  const float* WO = (const float*)d_in[4];
  float* out = (float*)d_out;

  char* ws = (char*)d_ws;
  _Float16* Qh  = (_Float16*)(ws);                              // 4096*2048 f16 = 16MB
  _Float16* Kh  = (_Float16*)(ws + (size_t)16 * 1024 * 1024);   // 4096*512  f16 =  4MB
  _Float16* Vh  = (_Float16*)(ws + (size_t)20 * 1024 * 1024);   //                 4MB
  _Float16* Vtp = (_Float16*)(ws + (size_t)24 * 1024 * 1024);   //                 4MB
  _Float16* ctx = (_Float16*)(ws + (size_t)28 * 1024 * 1024);   //                16MB

  // Q/K/V projections (f32 A, f16 out)
  gemm_wmma_kernel<<<dim3(D_MODEL/128, T_SEQ/128), 256, 0, stream>>>(x, WQ, Qh, T_SEQ, D_MODEL, D_MODEL);
  gemm_wmma_kernel<<<dim3(D_KV  /128, T_SEQ/128), 256, 0, stream>>>(x, WK, Kh, T_SEQ, D_KV,   D_MODEL);
  gemm_wmma_kernel<<<dim3(D_KV  /128, T_SEQ/128), 256, 0, stream>>>(x, WV, Vh, T_SEQ, D_KV,   D_MODEL);
  // RoPE on Q and K
  rope_kernel<<<(T_SEQ * NQH  * 64) / 256, 256, 0, stream>>>(Qh, NQH);
  rope_kernel<<<(T_SEQ * NKVH * 64) / 256, 256, 0, stream>>>(Kh, NKVH);
  // V -> dim-major
  transpose_v_kernel<<<(T_SEQ * D_KV) / 256, 256, 0, stream>>>(Vh, Vtp);
  // FlashAttention: 16 heads * 256 query blocks / 8 waves per block
  flash_attn_kernel<<<(NQH * (T_SEQ / 16)) / 8, 256, 0, stream>>>(Qh, Kh, Vtp, ctx);
  // Output projection (f16 A, f32 out)
  gemm_wmma_kernel<<<dim3(D_MODEL/128, T_SEQ/128), 256, 0, stream>>>(ctx, WO, out, T_SEQ, D_MODEL, D_MODEL);
}
